// AxialSelfAttentionModule_2482491097691
// MI455X (gfx1250) — matrix operand order repaired and checked by reading, not compiled, not run on hardware
//
#include <hip/hip_runtime.h>
#include <hip/hip_bf16.h>
#include <math.h>

// ---------------------------------------------------------------------------
// AxialSelfAttention for MI455X (gfx1250, wave32, WMMA).
// Math simplifications (exact in real arithmetic):
//  * pos_attn branch: pa[:, :, i, None] is constant per softmax row -> removed.
//  * rotation: R orthogonal and applied to BOTH q and k -> q_rot.k_rot = q.k.
//    (R6_d/h/w unused.)
//  * q_na is row-constant under softmax -> only 0.1*k_na[j] kept.
//  * qkv projection is axis-independent -> computed once, gathered 3x.
// All GEMMs (27-tap wrap conv as implicit GEMM, five 192x192 1x1 convs,
// qkv 192->576, proj) run on v_wmma_f32_16x16x32_f16 with f32 accumulation.
// ---------------------------------------------------------------------------

typedef _Float16 f16;
typedef _Float16 h8   __attribute__((ext_vector_type(8)));
typedef _Float16 v16h __attribute__((ext_vector_type(16)));
typedef float    v8f  __attribute__((ext_vector_type(8)));

#define BB   2
#define CC   192
#define SS   32768            // D*H*W
#define NN   (BB * SS)        // 65536 tokens
#define NHH  16
#define HDD  12
#define SCALE_ 0.2886751345948129f  // 12^-0.5

// ---------------- small utility kernels ----------------

__global__ void k_f32_to_f16(const float* __restrict__ in, f16* __restrict__ out, int n) {
  int i = blockIdx.x * 256 + threadIdx.x;
  if (i < n) out[i] = (f16)in[i];
}

// lp1_w [O=192][C=192][27] -> [tap][O][C] f16
__global__ void k_permute_lp1(const float* __restrict__ w, f16* __restrict__ out) {
  int i = blockIdx.x * 256 + threadIdx.x;
  if (i >= 192 * 192 * 27) return;
  int t = i % 27;
  int c = (i / 27) % 192;
  int o = i / (27 * 192);
  out[((size_t)t * 192 + o) * 192 + c] = (f16)w[i];
}

// [B,C,S] f32 -> [B,S,C] (f16 or f32), LDS-tiled for coalescing both sides
template <typename OUT>
__global__ void k_tok_transpose(const float* __restrict__ in, OUT* __restrict__ out) {
  __shared__ float tile[32][33];
  int b  = blockIdx.z;
  int c0 = blockIdx.y * 32;
  int s0 = blockIdx.x * 32;
  int tx = threadIdx.x, ty = threadIdx.y;   // 32 x 8
  #pragma unroll
  for (int r = 0; r < 32; r += 8)
    tile[ty + r][tx] = in[(size_t)b * CC * SS + (size_t)(c0 + ty + r) * SS + s0 + tx];
  __syncthreads();
  #pragma unroll
  for (int r = 0; r < 32; r += 8)
    out[((size_t)b * SS + s0 + ty + r) * CC + c0 + tx] = (OUT)tile[tx][ty + r];
}

// ---------------- WMMA fragment loaders (ISA 16-bit layouts) ----------------

__device__ __forceinline__ v16h load_a_frag(const f16* __restrict__ A, int K, int row,
                                            int kb, int lane) {
  int hf = lane >> 4;                       // A 16x32: two 8-half runs per lane
  const f16* p = A + (size_t)row * K + kb + hf * 8;
  h8 lo = *(const h8*)p;
  h8 hi = *(const h8*)(p + 16);
  v16h a;
  #pragma unroll
  for (int t = 0; t < 8; t++) { a[t] = lo[t]; a[8 + t] = hi[t]; }
  return a;
}

__device__ __forceinline__ v16h load_b_frag(const f16* __restrict__ W, int K, int col,
                                            int kb, int lane) {
  int hf = lane >> 4;                       // B 32x16: two 8-half runs per lane
  const f16* p = W + (size_t)col * K + kb + hf * 8;
  h8 lo = *(const h8*)p;
  h8 hi = *(const h8*)(p + 16);
  v16h b;
  #pragma unroll
  for (int t = 0; t < 8; t++) { b[t] = lo[t]; b[8 + t] = hi[t]; }
  return b;
}

// ---------------- generic WMMA GEMM: Y[N,O] = A[N,K] * W[O,K]^T + bias -----
// epi: 0 = f32 out, 1 = f16 out, 2 = sigmoid f32 out, 3 = store f32 to [B,C,S]

__global__ __launch_bounds__(256) void k_gemm(const f16* __restrict__ A,
                                              const f16* __restrict__ W,
                                              const float* __restrict__ bias,
                                              float* __restrict__ outF,
                                              f16* __restrict__ outH,
                                              int K, int O, int epi,
                                              float* __restrict__ bchw) {
  int wave = threadIdx.x >> 5, lane = threadIdx.x & 31;
  int mt = blockIdx.x * 8 + wave;           // token tile (16 tokens)
  int nt = blockIdx.y;                      // output tile (16 cols)
  int rA = mt * 16 + (lane & 15);
  int cB = nt * 16 + (lane & 15);
  v8f acc = {};
  for (int kb = 0; kb < K; kb += 32) {
    v16h a = load_a_frag(A, K, rA, kb, lane);
    v16h b = load_b_frag(W, K, cB, kb, lane);
    acc = __builtin_amdgcn_wmma_f32_16x16x32_f16(false, a, false, b, (short)0, acc,
                                                 false, false);
  }
  int hf  = lane >> 4;
  int col = nt * 16 + (lane & 15);
  float bv = bias ? bias[col] : 0.0f;
  #pragma unroll
  for (int v = 0; v < 8; v++) {
    int row = mt * 16 + v + hf * 8;
    float val = acc[v] + bv;
    if (epi == 0) {
      outF[(size_t)row * O + col] = val;
    } else if (epi == 1) {
      outH[(size_t)row * O + col] = (f16)val;
    } else if (epi == 2) {
      outF[(size_t)row * O + col] = 1.0f / (1.0f + expf(-val));
    } else {  // transpose-store epilogue to [B,C,S]
      int b_ = row >> 15, s = row & (SS - 1);
      bchw[(size_t)b_ * CC * SS + (size_t)col * SS + s] = val;
    }
  }
}

// ---------------- 3x3x3 wrap conv as implicit WMMA GEMM --------------------
// Pt: [N,192] f16 token-major pos_emb; W27: [27][O=192][C=192] f16.

__global__ __launch_bounds__(256) void k_conv3(const f16* __restrict__ Pt,
                                               const f16* __restrict__ W27,
                                               const float* __restrict__ bias,
                                               float* __restrict__ out) {
  int wave = threadIdx.x >> 5, lane = threadIdx.x & 31;
  int mt = blockIdx.x * 8 + wave;
  int nt = blockIdx.y;
  int tok = mt * 16 + (lane & 15);
  int b = tok >> 15, d = (tok >> 10) & 31, h = (tok >> 5) & 31, w = tok & 31;
  int cB = nt * 16 + (lane & 15);
  v8f acc = {};
  for (int t = 0; t < 27; t++) {
    int dz = t / 9 - 1, dy = (t / 3) % 3 - 1, dx = t % 3 - 1;
    int nb = (b << 15) | (((d + dz + 32) & 31) << 10) |
             (((h + dy + 32) & 31) << 5) | ((w + dx + 32) & 31);
    const f16* Wt = W27 + (size_t)t * 192 * 192;
    #pragma unroll
    for (int kb = 0; kb < 192; kb += 32) {
      v16h a = load_a_frag(Pt, 192, nb, kb, lane);
      v16h bf = load_b_frag(Wt, 192, cB, kb, lane);
      acc = __builtin_amdgcn_wmma_f32_16x16x32_f16(false, a, false, bf, (short)0, acc,
                                                   false, false);
    }
  }
  int hf = lane >> 4;
  int col = nt * 16 + (lane & 15);
  float bv = bias[col];
  #pragma unroll
  for (int v = 0; v < 8; v++)
    out[(size_t)(mt * 16 + v + hf * 8) * 192 + col] = acc[v] + bv;
}

// ---------------- instance-norm stats + fused inorm/gelu -------------------

__global__ void k_stats(const float* __restrict__ X, float* __restrict__ stats) {
  int c = threadIdx.x;
  if (c >= 192) return;
  int row0 = blockIdx.x * 256;
  float s = 0.0f, ss = 0.0f;
  for (int r = 0; r < 256; r++) {
    float v = X[(size_t)(row0 + r) * 192 + c];
    s += v; ss += v * v;
  }
  int b = row0 >> 15;
  atomicAdd(&stats[b * 384 + c], s);
  atomicAdd(&stats[b * 384 + 192 + c], ss);
}

__global__ void k_inorm_gelu(const float* __restrict__ X, const float* __restrict__ stats,
                             f16* __restrict__ out) {
  int i = blockIdx.x * 256 + threadIdx.x;
  if (i >= NN * 192) return;
  int c = i % 192;
  int b = (i / 192) >> 15;
  float m  = stats[b * 384 + c] * (1.0f / 32768.0f);
  float vr = stats[b * 384 + 192 + c] * (1.0f / 32768.0f) - m * m;
  float y  = (X[i] - m) * rsqrtf(vr + 1e-5f);
  out[i] = (f16)(0.5f * y * (1.0f + erff(y * 0.70710678118654752f)));  // exact gelu
}

// ---------------- modulation + vector-gate combine -> x_mod (f16) ----------

__global__ void k_combine(const float* __restrict__ Xt, const float* __restrict__ mod,
                          const float* __restrict__ vm, const float* __restrict__ vng_w,
                          const float* __restrict__ vng_b, f16* __restrict__ out) {
  int i = blockIdx.x * 256 + threadIdx.x;  // over NN * 64 vector groups
  if (i >= NN * 64) return;
  int vc = i % 64;
  size_t base = (size_t)(i / 64) * 192 + vc * 3;
  float x0 = Xt[base], x1 = Xt[base + 1], x2 = Xt[base + 2];
  float vn = sqrtf(x0 * x0 + x1 * x1 + x2 * x2);
  float gate = 1.0f / (1.0f + expf(-(vn * vng_w[vc] + vng_b[vc])));
  float m0 = vm[base], m1 = vm[base + 1], m2 = vm[base + 2];
  float inv = 1.0f / fmaxf(sqrtf(m0 * m0 + m1 * m1 + m2 * m2), 1e-8f);
  float g = gate * vn * inv;
  out[base]     = (f16)(x0 * mod[base]     + 0.1f * (x0 + g * m0));
  out[base + 1] = (f16)(x1 * mod[base + 1] + 0.1f * (x1 + g * m1));
  out[base + 2] = (f16)(x2 * mod[base + 2] + 0.1f * (x2 + g * m2));
}

// ---------------- axial attention (L=32 == wave32 lanes) -------------------
// One wave per (sequence, head). logits = SCALE*(q.k) + 0.1*k_na[j].

__global__ __launch_bounds__(128) void k_attn(const float* __restrict__ QKV,
                                              const float* __restrict__ vnp_w,
                                              const float* __restrict__ vnp_b,
                                              float* __restrict__ out,
                                              int strideTok, int loBits, int hiShift,
                                              int accumulate) {
  __shared__ float Ks[4][32][HDD];
  __shared__ float Vs[4][32][HDD];
  __shared__ float KNa[4][32];
  int wave = threadIdx.x >> 5, lane = threadIdx.x & 31;
  int p = blockIdx.x * 4 + wave;            // (seq, head) pair
  int q = p >> 4, h = p & 15;
  int base = ((q >> loBits) << hiShift) + (q & ((1 << loBits) - 1));
  int n = base + lane * strideTok;
  const float* qp = QKV + (size_t)n * 576 + h * HDD;
  float qi[HDD], kj[HDD];
  #pragma unroll
  for (int t = 0; t < HDD; t++) {
    qi[t] = qp[t];
    kj[t] = qp[192 + t];
    Ks[wave][lane][t] = kj[t];
    Vs[wave][lane][t] = qp[384 + t];
  }
  float kn = vnp_b[0];
  #pragma unroll
  for (int v = 0; v < 4; v++)
    kn += vnp_w[v] * sqrtf(kj[3 * v] * kj[3 * v] + kj[3 * v + 1] * kj[3 * v + 1] +
                           kj[3 * v + 2] * kj[3 * v + 2]);
  KNa[wave][lane] = kn;
  __syncthreads();

  float logits[32], mx = -1e30f;
  #pragma unroll 4
  for (int j = 0; j < 32; j++) {
    float dot = 0.0f;
    #pragma unroll
    for (int t = 0; t < HDD; t++) dot += qi[t] * Ks[wave][j][t];
    float l = dot * SCALE_ + 0.1f * KNa[wave][j];
    logits[j] = l;
    mx = fmaxf(mx, l);
  }
  float sum = 0.0f;
  #pragma unroll 4
  for (int j = 0; j < 32; j++) { float e = __expf(logits[j] - mx); logits[j] = e; sum += e; }
  float inv = 1.0f / sum;
  float o[HDD];
  #pragma unroll
  for (int t = 0; t < HDD; t++) o[t] = 0.0f;
  #pragma unroll 4
  for (int j = 0; j < 32; j++) {
    float wgt = logits[j] * inv;
    #pragma unroll
    for (int t = 0; t < HDD; t++) o[t] += wgt * Vs[wave][j][t];
  }
  float* op = out + (size_t)n * 192 + h * HDD;
  if (accumulate) {
    #pragma unroll
    for (int t = 0; t < HDD; t++) op[t] += o[t];
  } else {
    #pragma unroll
    for (int t = 0; t < HDD; t++) op[t] = o[t];
  }
}

// ---------------------------------------------------------------------------

extern "C" void kernel_launch(void* const* d_in, const int* in_sizes, int n_in,
                              void* d_out, int out_size, void* d_ws, size_t ws_size,
                              hipStream_t stream) {
  (void)in_sizes; (void)n_in; (void)out_size; (void)ws_size;
  const float* x       = (const float*)d_in[0];
  const float* pos_emb = (const float*)d_in[1];
  const float* qkv_w   = (const float*)d_in[2];
  const float* qkv_b   = (const float*)d_in[3];
  const float* lp1_w   = (const float*)d_in[4];
  const float* lp1_b   = (const float*)d_in[5];
  const float* lp2_w   = (const float*)d_in[6];
  const float* lp2_b   = (const float*)d_in[7];
  const float* vm1_w   = (const float*)d_in[8];
  const float* vm1_b   = (const float*)d_in[9];
  const float* vm2_w   = (const float*)d_in[10];
  const float* vm2_b   = (const float*)d_in[11];
  const float* md1_w   = (const float*)d_in[12];
  const float* md1_b   = (const float*)d_in[13];
  const float* md2_w   = (const float*)d_in[14];
  const float* md2_b   = (const float*)d_in[15];
  // d_in[16..17] pa_w/pa_b: softmax-invariant (dead). d_in[22..24] R6_*: rotation cancels.
  const float* vng_w   = (const float*)d_in[18];
  const float* vng_b   = (const float*)d_in[19];
  const float* vnp_w   = (const float*)d_in[20];
  const float* vnp_b   = (const float*)d_in[21];
  const float* proj_w  = (const float*)d_in[25];
  const float* proj_b  = (const float*)d_in[26];

  size_t cur = 0;
  char* ws = (char*)d_ws;
  auto alloc = [&](size_t bytes) -> void* {
    void* p = ws + cur;
    cur += (bytes + 255) & ~(size_t)255;
    return p;
  };
  f16*   wLp1p = (f16*)alloc((size_t)27 * 192 * 192 * 2);
  f16*   wLp2  = (f16*)alloc(36864 * 2);
  f16*   wMd1  = (f16*)alloc(36864 * 2);
  f16*   wMd2  = (f16*)alloc(36864 * 2);
  f16*   wVm1  = (f16*)alloc(36864 * 2);
  f16*   wVm2  = (f16*)alloc(36864 * 2);
  f16*   wProj = (f16*)alloc(36864 * 2);
  f16*   wQkv  = (f16*)alloc((size_t)576 * 192 * 2);
  float* stats = (float*)alloc(768 * 4);
  f16*   Ah    = (f16*)alloc((size_t)NN * 192 * 2);   // token-major pos_emb (f16)
  f16*   Eh    = (f16*)alloc((size_t)NN * 192 * 2);   // local (f16)
  f16*   Fh    = (f16*)alloc((size_t)NN * 192 * 2);   // gelu acts / x_mod / out (f16)
  float* Bf    = (float*)alloc((size_t)NN * 192 * 4); // GEMM f32 scratch
  float* Cf    = (float*)alloc((size_t)NN * 192 * 4); // mod
  float* Df    = (float*)alloc((size_t)NN * 192 * 4); // vm
  float* G     = (float*)alloc((size_t)NN * 192 * 4); // Xt, later out_sum
  float* QKV   = (float*)alloc((size_t)NN * 576 * 4);

  const int EW = (36864 + 255) / 256;
  // weights -> f16
  k_permute_lp1<<<(192 * 192 * 27 + 255) / 256, 256, 0, stream>>>(lp1_w, wLp1p);
  k_f32_to_f16<<<EW, 256, 0, stream>>>(lp2_w, wLp2, 36864);
  k_f32_to_f16<<<EW, 256, 0, stream>>>(md1_w, wMd1, 36864);
  k_f32_to_f16<<<EW, 256, 0, stream>>>(md2_w, wMd2, 36864);
  k_f32_to_f16<<<EW, 256, 0, stream>>>(vm1_w, wVm1, 36864);
  k_f32_to_f16<<<EW, 256, 0, stream>>>(vm2_w, wVm2, 36864);
  k_f32_to_f16<<<EW, 256, 0, stream>>>(proj_w, wProj, 36864);
  k_f32_to_f16<<<(110592 + 255) / 256, 256, 0, stream>>>(qkv_w, wQkv, 110592);

  dim3 tB(32, 8), tG(SS / 32, 6, BB);
  k_tok_transpose<f16><<<tG, tB, 0, stream>>>(pos_emb, Ah);
  k_tok_transpose<float><<<tG, tB, 0, stream>>>(x, G);

  dim3 g12(NN / 128, 12), g36(NN / 128, 36);
  const int gE = NN * 192 / 256;

  // local = conv1x1(gelu(inorm(circ_conv3(pos_emb))))
  k_conv3<<<g12, 256, 0, stream>>>(Ah, wLp1p, lp1_b, Bf);
  hipMemsetAsync(stats, 0, 768 * 4, stream);
  k_stats<<<NN / 256, 256, 0, stream>>>(Bf, stats);
  k_inorm_gelu<<<gE, 256, 0, stream>>>(Bf, stats, Fh);
  k_gemm<<<g12, 256, 0, stream>>>(Fh, wLp2, lp2_b, nullptr, Eh, 192, 192, 1, nullptr);

  // mod = sigmoid(conv1x1(gelu(inorm(conv1x1(local)))))
  k_gemm<<<g12, 256, 0, stream>>>(Eh, wMd1, md1_b, Bf, nullptr, 192, 192, 0, nullptr);
  hipMemsetAsync(stats, 0, 768 * 4, stream);
  k_stats<<<NN / 256, 256, 0, stream>>>(Bf, stats);
  k_inorm_gelu<<<gE, 256, 0, stream>>>(Bf, stats, Fh);
  k_gemm<<<g12, 256, 0, stream>>>(Fh, wMd2, md2_b, Cf, nullptr, 192, 192, 2, nullptr);

  // vm = conv1x1(gelu(inorm(conv1x1(local))))
  k_gemm<<<g12, 256, 0, stream>>>(Eh, wVm1, vm1_b, Bf, nullptr, 192, 192, 0, nullptr);
  hipMemsetAsync(stats, 0, 768 * 4, stream);
  k_stats<<<NN / 256, 256, 0, stream>>>(Bf, stats);
  k_inorm_gelu<<<gE, 256, 0, stream>>>(Bf, stats, Fh);
  k_gemm<<<g12, 256, 0, stream>>>(Fh, wVm2, vm2_b, Df, nullptr, 192, 192, 0, nullptr);

  // x_mod (f16), qkv once
  k_combine<<<NN * 64 / 256, 256, 0, stream>>>(G, Cf, Df, vng_w, vng_b, Fh);
  k_gemm<<<g36, 256, 0, stream>>>(Fh, wQkv, qkv_b, QKV, nullptr, 192, 576, 0, nullptr);

  // three axial attentions accumulate into G (reused as out_sum)
  k_attn<<<8192, 128, 0, stream>>>(QKV, vnp_w, vnp_b, G, 1024, 10, 15, 0); // depth
  k_attn<<<8192, 128, 0, stream>>>(QKV, vnp_w, vnp_b, G, 32, 5, 10, 1);    // height
  k_attn<<<8192, 128, 0, stream>>>(QKV, vnp_w, vnp_b, G, 1, 0, 5, 1);      // width

  // final proj, epilogue stores transposed [B,C,D,H,W] f32 into d_out
  k_f32_to_f16<<<gE, 256, 0, stream>>>(G, Fh, NN * 192);
  k_gemm<<<g12, 256, 0, stream>>>(Fh, wProj, proj_b, nullptr, nullptr, 192, 192, 3,
                                  (float*)d_out);
}
